// CARAFEpp_68204080660808
// MI455X (gfx1250) — matrix operand order repaired and checked by reading, not compiled, not run on hardware
//
#include <hip/hip_runtime.h>

typedef __attribute__((ext_vector_type(16))) _Float16 v16h;
typedef __attribute__((ext_vector_type(8)))  _Float16 v8h;
typedef __attribute__((ext_vector_type(8)))  float    v8f;

#define HWp   4096     // 64*64
#define CIN   256
#define MIDc  64
#define ENCK  576      // 64 * 9
#define ENCMP 112      // 100 padded to multiple of 16
#define XSTR  264      // compress LDS pixel-row stride (halfs): 16B-aligned, bank-stride 4
#define PSTR  584      // encoder patch-row stride (halfs): 16B-aligned, bank-stride 36

// Load a 16-half fragment slice as two 16-byte LDS reads.
__device__ __forceinline__ v16h lds_frag16(const _Float16* p) {
    v8h lo = *(const v8h*)p;
    v8h hi = *(const v8h*)(p + 8);
    v16h r;
    #pragma unroll
    for (int i = 0; i < 8; ++i) { r[i] = lo[i]; r[i + 8] = hi[i]; }
    return r;
}

// ---------------------------------------------------------------------------
// Kernel 0: weight conversion fp32 -> f16 (zero-pad encoder M to 112)
// ---------------------------------------------------------------------------
__global__ __launch_bounds__(256) void carafe_prep(
    const float* __restrict__ Wcomp, const float* __restrict__ Wenc,
    _Float16* __restrict__ Wc16, _Float16* __restrict__ We16)
{
    int idx = blockIdx.x * 256 + threadIdx.x;
    if (idx < MIDc * CIN) {
        Wc16[idx] = (_Float16)Wcomp[idx];
    }
    int j = idx - MIDc * CIN;
    if (j >= 0 && j < ENCMP * ENCK) {
        // W_enc flat layout (100,64,3,3) == m*576 + c*9 + r : already our K order
        We16[j] = (j < 100 * ENCK) ? (_Float16)Wenc[j] : (_Float16)0.f;
    }
}

// ---------------------------------------------------------------------------
// Kernel 1: 1x1 compress conv (GEMM 64x256 * 256xNpix) + bias + ReLU -> f16
// LDS tile is pixel-major so each B fragment is 2x ds_load_b128.
// ---------------------------------------------------------------------------
__global__ __launch_bounds__(256) void carafe_compress(
    const float* __restrict__ x, const _Float16* __restrict__ Wc16,
    const float* __restrict__ bcomp, _Float16* __restrict__ feat)
{
    __shared__ __align__(16) _Float16 xs[32 * XSTR];   // [pix][c]
    int wg   = blockIdx.x;                             // 512 WGs
    int b    = wg >> 7;
    int pos0 = (wg & 127) << 5;
    int tid  = threadIdx.x;

    const float* xb = x + (size_t)b * CIN * HWp + pos0;
    #pragma unroll
    for (int it = 0; it < 8; ++it) {
        int e  = tid + (it << 8);                      // 2048 float4 units
        int c  = e >> 3;
        int p4 = (e & 7) << 2;
        float4 v = *(const float4*)(xb + (size_t)c * HWp + p4);
        xs[(p4 + 0) * XSTR + c] = (_Float16)v.x;
        xs[(p4 + 1) * XSTR + c] = (_Float16)v.y;
        xs[(p4 + 2) * XSTR + c] = (_Float16)v.z;
        xs[(p4 + 3) * XSTR + c] = (_Float16)v.w;
    }
    __syncthreads();

    int lane = tid & 31, wid = tid >> 5;
    int m0 = (wid & 3) << 4;                           // M tile 0..3
    int p0 = (wid >> 2) << 4;                          // pixel tile 0..1
    int lrow = lane & 15;
    bool hi = (lane >= 16);

    const _Float16* arow = Wc16 + (m0 + lrow) * CIN;
    const _Float16* brow = &xs[(p0 + lrow) * XSTR];

    v8f acc = {};
    #pragma unroll
    for (int k0 = 0; k0 < CIN; k0 += 32) {
        v16h a;
        int kba = k0 + (hi ? 16 : 0);                  // half-split-K A layout
        #pragma unroll
        for (int i = 0; i < 16; ++i)
            a[i] = arow[kba + i];
        int kbb = k0 + (hi ? 16 : 0);                  // half-split-K B layout
        v16h bb = lds_frag16(brow + kbb);
        acc = __builtin_amdgcn_wmma_f32_16x16x32_f16(
                  false, a, false, bb, (short)0, acc, false, false);
    }

    int mbase = m0 + (hi ? 8 : 0);
    int pixc  = pos0 + p0 + lrow;
    _Float16* fb = feat + (size_t)b * MIDc * HWp + pixc;
    #pragma unroll
    for (int r = 0; r < 8; ++r) {
        float v = acc[r] + bcomp[mbase + r];
        v = v > 0.f ? v : 0.f;
        fb[(size_t)(mbase + r) * HWp] = (_Float16)v;
    }
}

// ---------------------------------------------------------------------------
// Kernel 2: 3x3 encoder conv (im2col GEMM K=576, M=112, N=32 pixels/WG)
//           + bias + softmax over 25 taps. im2col expanded once into a packed
//           patch matrix P so B fragments are 2x ds_load_b128.
// ---------------------------------------------------------------------------
__global__ __launch_bounds__(256) void carafe_encoder(
    const _Float16* __restrict__ feat, const _Float16* __restrict__ We16,
    const float* __restrict__ benc, float* __restrict__ wk)
{
    __shared__ union SU {
        _Float16 fs[MIDc * 3 * 36];                    // halo: dead after expansion
        float    wkbuf[ENCMP * 33];                    // GEMM output staging
    } u;
    __shared__ __align__(16) _Float16 P[32 * PSTR];    // [pix][K] packed patches

    int wg  = blockIdx.x;                              // 512 = B * H * (W/32)
    int b   = wg >> 7;
    int rem = wg & 127;
    int h   = rem >> 1;
    int w0  = (rem & 1) << 5;
    int tid = threadIdx.x;

    // phase 1: stage 64ch x 3row x 34col halo (zero padded)
    const _Float16* fbase = feat + (size_t)b * MIDc * HWp;
    for (int e = tid; e < MIDc * 3 * 34; e += 256) {
        int c  = e / 102;
        int r2 = e - c * 102;
        int di = r2 / 34;
        int cc = r2 - di * 34;
        int row = h + di - 1;
        int col = w0 + cc - 1;
        _Float16 v = (_Float16)0.f;
        if (row >= 0 && row < 64 && col >= 0 && col < 64)
            v = fbase[(size_t)c * HWp + row * 64 + col];
        u.fs[(c * 3 + di) * 36 + cc] = v;
    }
    __syncthreads();

    // phase 2: expand im2col patches: P[pix][K], K = c*9 + di*3 + dj.
    // (channel, tap) loop with compile-time tap decomposition: only
    // immediate-offset ds ops, no integer divisions.
    {
        int pix = tid >> 3;                            // 32 pixels x 8 threads
        int c0  = (tid & 7) << 3;                      // 8 channels per thread
        #pragma unroll
        for (int cc = 0; cc < 8; ++cc) {
            int c = c0 + cc;
            const _Float16* src = &u.fs[c * 108 + pix];   // (c*3+di)*36 + pix + dj
            _Float16* dst = &P[pix * PSTR + c * 9];
            #pragma unroll
            for (int r = 0; r < 9; ++r) {
                int di = r / 3, dj = r - (r / 3) * 3;  // compile-time constants
                dst[r] = src[di * 36 + dj];
            }
        }
    }
    __syncthreads();

    // phase 3: GEMM, 14 (M,N) tiles over 8 waves, 18 WMMAs each
    int lane = tid & 31, wid = tid >> 5;
    int lrow = lane & 15;
    bool hi = (lane >= 16);

    for (int t = wid; t < 14; t += 8) {
        int m0 = (t % 7) << 4;
        int n0 = (t / 7) << 4;
        const _Float16* arow = We16 + (m0 + lrow) * ENCK;
        const _Float16* brow = &P[(n0 + lrow) * PSTR];
        int wl = n0 + lrow;
        v8f acc = {};
        #pragma unroll
        for (int k0 = 0; k0 < ENCK; k0 += 32) {
            v16h a;
            int kba = k0 + (hi ? 16 : 0);
            #pragma unroll
            for (int i = 0; i < 16; ++i)
                a[i] = arow[kba + i];
            int kbb = k0 + (hi ? 16 : 0);
            v16h bb = lds_frag16(brow + kbb);
            acc = __builtin_amdgcn_wmma_f32_16x16x32_f16(
                      false, a, false, bb, (short)0, acc, false, false);
        }
        int mbase = m0 + (hi ? 8 : 0);
        #pragma unroll
        for (int r = 0; r < 8; ++r) {
            int mr = mbase + r;
            float bias = (mr < 100) ? benc[mr] : 0.f;
            u.wkbuf[mr * 33 + wl] = acc[r] + bias;     // safe: fs dead, P untouched
        }
    }
    __syncthreads();

    // phase 4: softmax over 25 taps: 32 pixels x 4 sub-positions
    if (tid < 128) {
        int wl = tid & 31;
        int p  = tid >> 5;
        float vals[25];
        float mx = -1e30f;
        #pragma unroll
        for (int k = 0; k < 25; ++k) {
            vals[k] = u.wkbuf[(p * 25 + k) * 33 + wl];
            mx = fmaxf(mx, vals[k]);
        }
        float s = 0.f;
        #pragma unroll
        for (int k = 0; k < 25; ++k) {
            vals[k] = __expf(vals[k] - mx);
            s += vals[k];
        }
        float inv = 1.f / s;
        int pos = h * 64 + w0 + wl;
        float* o = wk + ((size_t)(b * HWp + pos) * 100 + p * 25);
        #pragma unroll
        for (int k = 0; k < 25; ++k) o[k] = vals[k] * inv;
    }
}

// ---------------------------------------------------------------------------
// Kernel 3: CARAFE gather + pixel shuffle. WG per (b,h) row, 64w x 4ch threads.
// Per-pixel 100 softmax weights live in registers across the 64 ch-groups.
// out flat = (b*C+c)*16384 + h*4W + 4w + p  (the view/permute/view shuffle)
// ---------------------------------------------------------------------------
__global__ __launch_bounds__(256) void carafe_gather(
    const float* __restrict__ x, const float* __restrict__ wk,
    float* __restrict__ out)
{
    __shared__ float xs[4 * 5 * 72];                   // 4 ch x 5 rows x 68(+4) cols
    int wg  = blockIdx.x;                              // 256 = B * H
    int b   = wg >> 6;
    int h   = wg & 63;
    int tid = threadIdx.x;
    int w   = tid & 63;
    int ci  = tid >> 6;

    // 100 weights per pixel, 400B-aligned -> 25x global_load_b128
    float wkr[100];
    const float4* wp4 = (const float4*)(wk + (size_t)(b * HWp + h * 64 + w) * 100);
    #pragma unroll
    for (int q = 0; q < 25; ++q) {
        float4 t = wp4[q];
        wkr[4 * q + 0] = t.x;
        wkr[4 * q + 1] = t.y;
        wkr[4 * q + 2] = t.z;
        wkr[4 * q + 3] = t.w;
    }

    const float* xb = x + (size_t)b * CIN * HWp;
    float* ob = out + (size_t)b * CIN * 16384 + h * 256 + w * 4;

    for (int cg = 0; cg < 64; ++cg) {
        int cbase = cg << 2;
        __syncthreads();
        for (int e = tid; e < 4 * 5 * 68; e += 256) {
            int c2  = e / 340;
            int rr  = e - c2 * 340;
            int di  = rr / 68;
            int col = rr - di * 68;
            int row = h + di - 2;
            int xc  = col - 2;
            float v = 0.f;
            if (row >= 0 && row < 64 && xc >= 0 && xc < 64)
                v = xb[(size_t)(cbase + c2) * HWp + row * 64 + xc];
            xs[(c2 * 5 + di) * 72 + col] = v;
        }
        __syncthreads();

        if (cg < 63)                                    // pull next group toward L2/L0
            __builtin_prefetch(xb + (size_t)(cbase + 4 + ci) * HWp + h * 64 + w, 0, 1);

        float a0 = 0.f, a1 = 0.f, a2 = 0.f, a3 = 0.f;
        #pragma unroll
        for (int k = 0; k < 25; ++k) {
            int di = k / 5, dj = k - (k / 5) * 5;
            float xv = xs[(ci * 5 + di) * 72 + w + dj];
            a0 += wkr[k]      * xv;
            a1 += wkr[25 + k] * xv;
            a2 += wkr[50 + k] * xv;
            a3 += wkr[75 + k] * xv;
        }
        float4 o4 = make_float4(a0, a1, a2, a3);
        *(float4*)(ob + (size_t)(cbase + ci) * 16384) = o4;
    }
}

// ---------------------------------------------------------------------------
extern "C" void kernel_launch(void* const* d_in, const int* in_sizes, int n_in,
                              void* d_out, int out_size, void* d_ws, size_t ws_size,
                              hipStream_t stream) {
    const float* x     = (const float*)d_in[0];
    const float* Wcomp = (const float*)d_in[1];
    const float* bcomp = (const float*)d_in[2];
    const float* Wenc  = (const float*)d_in[3];
    const float* benc  = (const float*)d_in[4];
    float* out = (float*)d_out;

    char* ws = (char*)d_ws;
    _Float16* Wc16 = (_Float16*)(ws);                               // 32 KB
    _Float16* We16 = (_Float16*)(ws + 32768);                       // 126 KB
    _Float16* feat = (_Float16*)(ws + 32768 + 129024);              // 2 MB
    float*    wkg  = (float*)   (ws + 32768 + 129024 + 2097152);    // 6.25 MB

    carafe_prep    <<<316, 256, 0, stream>>>(Wcomp, Wenc, Wc16, We16);
    carafe_compress<<<512, 256, 0, stream>>>(x, Wc16, bcomp, feat);
    carafe_encoder <<<512, 256, 0, stream>>>(feat, We16, benc, wkg);
    carafe_gather  <<<256, 256, 0, stream>>>(x, wkg, out);
}
